// EFDeepSet_34574486733481
// MI455X (gfx1250) — matrix operand order repaired and checked by reading, not compiled, not run on hardware
//
#include <hip/hip_runtime.h>
#include <math.h>

typedef __attribute__((ext_vector_type(16))) _Float16 v16h;
typedef __attribute__((ext_vector_type(8)))  _Float16 v8h;
typedef __attribute__((ext_vector_type(8)))  float    v8f;

// ---------------------------------------------------------------------------
// A-fragment: lane holds row M = lane&15.
// Element e -> K = (e&7) + 8*(lane>=16) + 16*(e>=8)  (ISA 7.12.2, 16-bit A 16x32)
// => two contiguous 16B chunks per lane from LDS.
__device__ __forceinline__ v16h load_a_frag(const _Float16* row, int kbase, int laneHi) {
  v8h lo = *(const v8h*)(row + kbase + laneHi * 8);
  v8h hi = *(const v8h*)(row + kbase + 16 + laneHi * 8);
  return __builtin_shufflevector(lo, hi, 0,1,2,3,4,5,6,7,8,9,10,11,12,13,14,15);
}

// C/D layout: VGPR r -> row (r + 8*laneHi), col = lane&15 (ISA 7.12.2).
__device__ __forceinline__ void store_c_relu(const v8f& acc, _Float16* obase,
                                             int rowBase, int stride, int laneHi) {
#pragma unroll
  for (int r = 0; r < 8; ++r) {
    float v = fmaxf(acc[r], 0.0f);                 // fused ReLU
    obase[(rowBase + r + 8 * laneHi) * stride] = (_Float16)v;
  }
}

// One dense layer pass over a 32-row LDS tile: out = relu(in @ Wpacked).
// nt-outer / kt-inner: each B fragment (weights, global) is loaded ONCE and
// feeds both 16-row tiles -> 2 WMMAs per B load, half the weight traffic.
// wpack layout: [kt][nt][lane][16 halves] (each lane loads 32B contiguous).
template <int NT, int KT>
__device__ __forceinline__ void wmma_layer(const _Float16* in, int inStride,
                                           const _Float16* wpack,
                                           _Float16* out, int outStride,
                                           int lane, int wave) {
  const int m = lane & 15;
  const int laneHi = lane >> 4;
  const _Float16* arow0 = in + m * inStride;            // row tile 0
  const _Float16* arow1 = in + (16 + m) * inStride;     // row tile 1
  for (int nt = wave; nt < NT; nt += 8) {               // 8 waves / 256-thread WG
    v8f acc0 = {0.f, 0.f, 0.f, 0.f, 0.f, 0.f, 0.f, 0.f};
    v8f acc1 = {0.f, 0.f, 0.f, 0.f, 0.f, 0.f, 0.f, 0.f};
#pragma unroll
    for (int kt = 0; kt < KT; ++kt) {
      v16h b = *(const v16h*)(wpack + (((kt * NT + nt) * 32 + lane) << 4));
      v16h a0 = load_a_frag(arow0, kt * 32, laneHi);
      acc0 = __builtin_amdgcn_wmma_f32_16x16x32_f16(false, a0, false, b,
                                                    (short)0, acc0, false, false);
      v16h a1 = load_a_frag(arow1, kt * 32, laneHi);
      acc1 = __builtin_amdgcn_wmma_f32_16x16x32_f16(false, a1, false, b,
                                                    (short)0, acc1, false, false);
    }
    _Float16* obase = out + nt * 16 + m;
    store_c_relu(acc0, obase, 0, outStride, laneHi);
    store_c_relu(acc1, obase, 16, outStride, laneHi);
  }
}

// ---------------------------------------------------------------------------
// Pack a row-major f32 [K,N] weight matrix into the f16 B-fragment layout:
// dst[((kt*(N/16)+nt)*32 + lane)*16 + e] = src[(kt*32 + 8*(lane>>4) + (e&7) + 16*(e>>3))*N + nt*16 + (lane&15)]
__global__ void pack_b_f16(const float* __restrict__ src, _Float16* __restrict__ dst,
                           int K, int N) {
  int p = blockIdx.x * 256 + threadIdx.x;
  if (p >= K * N) return;
  int e    = p & 15;
  int lane = (p >> 4) & 31;
  int tid  = p >> 9;
  int nTiles = N >> 4;
  int nt = tid % nTiles;
  int kt = tid / nTiles;
  int k = kt * 32 + ((lane >> 4) << 3) + (e & 7) + ((e >> 3) << 4);
  int n = nt * 16 + (lane & 15);
  dst[p] = (_Float16)src[k * N + n];
}

// ---------------------------------------------------------------------------
// Node encoder: h = relu(relu(relu(normalize(x)@W1)@W2)@W3), stored f16.
// 32 nodes per 256-thread block. Layer1 (K=14) in VALU f32, layers 2/3 WMMA.
__global__ void node_encoder(const float* __restrict__ x,
                             const float* __restrict__ W1,
                             const _Float16* __restrict__ W2p,
                             const _Float16* __restrict__ W3p,
                             _Float16* __restrict__ hout, int Nn) {
  __shared__ float sx[32][16];
  __shared__ float sinv[32];
  __shared__ __align__(16) _Float16 t1[32 * 136];
  __shared__ __align__(16) _Float16 t2[32 * 136];
  const int t = threadIdx.x;
  const int lane = t & 31, wave = t >> 5;
  const int n0 = blockIdx.x * 32;

  for (int i = t; i < 32 * 14; i += 256) {
    int r = i / 14, c = i % 14;
    int n = n0 + r; if (n >= Nn) n = Nn - 1;
    sx[r][c] = x[n * 14 + c];
  }
  __syncthreads();
  if (t < 32) {
    float s = 0.f;
    for (int c = 0; c < 14; ++c) s += sx[t][c] * sx[t][c];
    sinv[t] = 1.0f / fmaxf(sqrtf(s), 1e-12f);
  }
  __syncthreads();
  // layer 1 (VALU): 32x128 outputs, 16 per thread
  {
    const int r = t >> 3;
    const int c0 = (t & 7) * 16;
    const float rcp = sinv[r];
#pragma unroll 4
    for (int i = 0; i < 16; ++i) {
      int c = c0 + i;
      float acc = 0.f;
      for (int k = 0; k < 14; ++k) acc = fmaf(sx[r][k] * rcp, W1[k * 128 + c], acc);
      t1[r * 136 + c] = (_Float16)fmaxf(acc, 0.f);
    }
  }
  __syncthreads();
  wmma_layer<8, 4>(t1, 136, W2p, t2, 136, lane, wave);
  __syncthreads();
  wmma_layer<8, 4>(t2, 136, W3p, t1, 136, lane, wave);
  __syncthreads();
  {
    const int r = t >> 3;
    const int c0 = (t & 7) * 16;
    const int n = n0 + r;
    if (n < Nn) {
      v8h* dst = (v8h*)(hout + (size_t)n * 128 + c0);
      dst[0] = *(const v8h*)(t1 + r * 136 + c0);
      dst[1] = *(const v8h*)(t1 + r * 136 + c0 + 8);
    }
  }
}

// ---------------------------------------------------------------------------
// Edge MLP: gather h[src],h[dst], invariant=[|xi-xj|, xi+xj] (E x 256),
// two WMMA layers 256->256 with ReLU, then 256->1 dot + scaled sigmoid.
// 32 edges per 256-thread block.
__global__ void edge_mlp(const _Float16* __restrict__ h,
                         const int* __restrict__ ei, int E,
                         const _Float16* __restrict__ A1p,
                         const _Float16* __restrict__ A2p,
                         const float* __restrict__ A3,
                         float* __restrict__ out) {
  __shared__ __align__(16) _Float16 b0[32 * 264];  // invariant, later a2
  __shared__ __align__(16) _Float16 b1[32 * 264];  // a1
  const int t = threadIdx.x;
  const int lane = t & 31, wave = t >> 5;
  const int e0 = blockIdx.x * 32;

  // gather + build permutation invariant (16 cols of H per thread)
  {
    const int r = t >> 3;
    const int c0 = (t & 7) * 16;
    int e = e0 + r; if (e >= E) e = E - 1;
    const int si = ei[e];
    const int di = ei[E + e];
    const v8h* pi = (const v8h*)(h + (size_t)si * 128 + c0);
    const v8h* pj = (const v8h*)(h + (size_t)di * 128 + c0);
    v8h iA = pi[0], iB = pi[1];
    v8h jA = pj[0], jB = pj[1];
    _Float16* row = b0 + r * 264;
#pragma unroll
    for (int i = 0; i < 8; ++i) {
      _Float16 d = iA[i] - jA[i]; if (d < (_Float16)0.f) d = -d;
      row[c0 + i]           = d;
      row[128 + c0 + i]     = iA[i] + jA[i];
      d = iB[i] - jB[i]; if (d < (_Float16)0.f) d = -d;
      row[c0 + 8 + i]       = d;
      row[128 + c0 + 8 + i] = iB[i] + jB[i];
    }
  }
  __syncthreads();
  wmma_layer<16, 8>(b0, 264, A1p, b1, 264, lane, wave);
  __syncthreads();
  wmma_layer<16, 8>(b1, 264, A2p, b0, 264, lane, wave);
  __syncthreads();
  // final 256->1 dot: 8 threads per edge, 32 elements each, shuffle-reduce
  {
    const int r = t >> 3;
    const int j = t & 7;
    const _Float16* row = b0 + r * 264;
    float acc = 0.f;
#pragma unroll 8
    for (int k = j * 32; k < j * 32 + 32; ++k) acc = fmaf((float)row[k], A3[k], acc);
    acc += __shfl_xor(acc, 1, 32);
    acc += __shfl_xor(acc, 2, 32);
    acc += __shfl_xor(acc, 4, 32);
    if (j == 0) {
      int e = e0 + r;
      if (e < E) {
        float sig = 1.0f / (1.0f + __expf(-acc));
        out[e] = 1e-8f + (1.0f - 2e-8f) * sig;
      }
    }
  }
}

// ---------------------------------------------------------------------------
extern "C" void kernel_launch(void* const* d_in, const int* in_sizes, int n_in,
                              void* d_out, int out_size, void* d_ws, size_t ws_size,
                              hipStream_t stream) {
  const float* x   = (const float*)d_in[0];
  const int*   ei  = (const int*)d_in[1];
  const float* W1  = (const float*)d_in[2];
  const float* W2  = (const float*)d_in[3];
  const float* W3  = (const float*)d_in[4];
  const float* A1  = (const float*)d_in[5];
  const float* A2  = (const float*)d_in[6];
  const float* A3  = (const float*)d_in[7];
  float* out = (float*)d_out;

  const int Nn = in_sizes[0] / 14;   // 50000
  const int E  = in_sizes[1] / 2;    // 800000

  // workspace carve-out (all 256B aligned)
  char* ws = (char*)d_ws;
  size_t off = 0;
  _Float16* hf = (_Float16*)(ws + off);
  off += ((size_t)Nn * 128 * 2 + 255) & ~(size_t)255;
  _Float16* A1p = (_Float16*)(ws + off); off += 256 * 256 * 2;
  _Float16* A2p = (_Float16*)(ws + off); off += 256 * 256 * 2;
  _Float16* W2p = (_Float16*)(ws + off); off += 128 * 128 * 2;
  _Float16* W3p = (_Float16*)(ws + off); off += 128 * 128 * 2;
  (void)ws_size; (void)n_in; (void)out_size;

  // 1) pack weights into WMMA B-fragment layout (f16)
  pack_b_f16<<<(256 * 256 + 255) / 256, 256, 0, stream>>>(A1, A1p, 256, 256);
  pack_b_f16<<<(256 * 256 + 255) / 256, 256, 0, stream>>>(A2, A2p, 256, 256);
  pack_b_f16<<<(128 * 128 + 255) / 256, 256, 0, stream>>>(W2, W2p, 128, 128);
  pack_b_f16<<<(128 * 128 + 255) / 256, 256, 0, stream>>>(W3, W3p, 128, 128);

  // 2) node encoder -> h (f16, [N,128])
  node_encoder<<<(Nn + 31) / 32, 256, 0, stream>>>(x, W1, W2p, W3p, hf, Nn);

  // 3) edge MLP -> out [E]
  edge_mlp<<<(E + 31) / 32, 256, 0, stream>>>(hf, ei, E, A1p, A2p, A3, out);
}
